// U_GCN_721554506463
// MI455X (gfx1250) — matrix operand order repaired and checked by reading, not compiled, not run on hardware
//
#include <hip/hip_runtime.h>
#include <hip/hip_bf16.h>

// ---------------------------------------------------------------------------
// CDNA5 (gfx1250) GAT implementation, round 2.
// Roofline: ~25 GF of matmul (all on v_wmma_f32_16x16x32_f16, f32 accum) vs
// ~1 GB of dense-mask HBM traffic -> bandwidth bound on the mask stream.
// B tiles (Wh, f16-transposed) are staged into LDS via the Tensor Data Mover
// (tensor_load_to_lds + s_wait_tensorcnt), double-buffered, shared by all
// 8 waves of the workgroup. Mask/er reads are float4 (global_load_b128) with
// global_prefetch_b8 lookahead.
// ---------------------------------------------------------------------------

typedef __attribute__((ext_vector_type(16))) _Float16 v16h;
typedef __attribute__((ext_vector_type(8)))  float    v8f;
typedef __attribute__((ext_vector_type(4)))  unsigned int u32x4;
typedef __attribute__((ext_vector_type(8)))  int      i32x8;
typedef __attribute__((ext_vector_type(4)))  int      i32x4;
typedef unsigned int       u32;
typedef unsigned long long u64;

#define N_NODES 4096
#define FOUT    64

// LDS tile: 64 rows (n) x 32 halves (k), row stride 64B + 4B TDM pad = 68B
#define TILE_ROW_BYTES 68
#define TILE_LDS (64 * TILE_ROW_BYTES)   // 4352 B per buffer

#if defined(__has_builtin)
#  if __has_builtin(__builtin_amdgcn_tensor_load_to_lds)
#    define HAVE_TDM 1
#  endif
#endif
#ifndef HAVE_TDM
#  define HAVE_TDM 0
#endif

// --- WMMA wrapper (codegen-confirmed builtin) ------------------------------
__device__ __forceinline__ v8f wmma_f16(v16h a, v16h b, v8f c) {
    return __builtin_amdgcn_wmma_f32_16x16x32_f16(
        false, a, false, b, (short)0, c, false, false);
}

// ---------------------------------------------------------------------------
// TDM: DMA a 64(n) x 32(k) f16 tile from WhT (row-major [64][4096], f16) into
// LDS with 1-DWORD pad per 64B row.  D# per ISA ch.8 (2D tensor, groups 2/3
// zero).  Issued by wave 0 only; completion via s_wait_tensorcnt + barrier.
// ---------------------------------------------------------------------------
#if HAVE_TDM
__device__ __forceinline__ void tdm_load_tile(const _Float16* gsrc, u32 lds_addr) {
    u64 ga = (u64)(uintptr_t)gsrc;
    u32x4 g0;
    g0[0] = 1u;                                   // count=1, user desc
    g0[1] = lds_addr;                             // LDS byte address
    g0[2] = (u32)ga;                              // global_addr[31:0]
    g0[3] = ((u32)(ga >> 32) & 0x01FFFFFFu)       // global_addr[56:32]
            | 0x80000000u;                        // type=2 ("image")
    i32x8 g1;
    g1[0] = 0x00D10000;   // data_size=2B | pad_enable | pad_interval=64B | pad=1dw
    g1[1] = (int)0x10000000;  // tensor_dim0 = 4096 (bits 79:48, low16 here)
    g1[2] = 0x00400000;   // tensor_dim1 = 64  (bits 111:80)
    g1[3] = 0x00200000;   // tile_dim0   = 32  (bits 127:112)
    g1[4] = 64;           // tile_dim1   = 64  (bits 143:128)
    g1[5] = 4096;         // tensor_dim0_stride = 4096 (bits 207:160)
    g1[6] = 0;
    g1[7] = 0;
    i32x4 z4 = {0, 0, 0, 0};
#if __clang_major__ >= 23
    i32x8 z8 = {0, 0, 0, 0, 0, 0, 0, 0};
    __builtin_amdgcn_tensor_load_to_lds(g0, g1, z4, z4, z8, 0);
#else
    __builtin_amdgcn_tensor_load_to_lds(g0, g1, z4, z4, 0);
#endif
}
#endif

// Stage one K-chunk tile; TDM path = wave0 DMA, fallback = cooperative copy.
__device__ __forceinline__ void stage_tile(const _Float16* gsrc, char* lbuf,
                                           u32 lds_addr, int tid) {
#if HAVE_TDM
    if (tid < 32) tdm_load_tile(gsrc, lds_addr);
    (void)lbuf;
#else
    for (int t = tid; t < 64 * 4; t += 256) {
        int n = t & 63, seg = t >> 6;              // 4 x 16B per 64B row
        const u32* s = (const u32*)(gsrc + (size_t)n * N_NODES + seg * 8);
        u32* d = (u32*)(lbuf + n * TILE_ROW_BYTES + seg * 16);
        d[0] = s[0]; d[1] = s[1]; d[2] = s[2]; d[3] = s[3];
    }
    (void)lds_addr;
#endif
}

__device__ __forceinline__ void tdm_wait(int pending, int tid) {
#if HAVE_TDM
    if (tid < 32) {
        if (pending) __builtin_amdgcn_s_wait_tensorcnt(1);
        else         __builtin_amdgcn_s_wait_tensorcnt(0);
    }
#else
    (void)pending; (void)tid;
#endif
}

// Read a 32x16 f16 B fragment (this lane: col n, K = half*16 + e) from the
// padded LDS tile; 8 DWORD reads, 17-DWORD lane stride -> conflict-free.
__device__ __forceinline__ v16h lds_frag(const char* p) {
    union { v16h h; u32 u[8]; } r;
#pragma unroll
    for (int d = 0; d < 8; ++d) r.u[d] = ((const u32*)p)[d];
    return r.h;
}

// ---------------------------------------------------------------------------
// Kernel 1: C[M,64] = A[M,K] @ B[K,64]  (f32 in/out, f16 WMMA, f32 accum)
// ---------------------------------------------------------------------------
#define CVT4(dst, B, v) \
    dst[B + 0] = (_Float16)(v).x; dst[B + 1] = (_Float16)(v).y; \
    dst[B + 2] = (_Float16)(v).z; dst[B + 3] = (_Float16)(v).w;

__global__ void gemm_k(const float* __restrict__ A, const float* __restrict__ B,
                       float* __restrict__ C, int K, int lda) {
    const int wave = threadIdx.x >> 5;
    const int lane = threadIdx.x & 31;
    const int half = (lane >> 4) & 1;
    const int lm   = lane & 15;
    const int row0 = (blockIdx.x * 8 + wave) * 16;

    v8f acc[4] = {};
    const float* arow = A + (size_t)(row0 + lm) * lda;

    for (int k0 = 0; k0 < K; k0 += 32) {
        const float4* a4 = (const float4*)(arow + k0);
        float4 p0 = a4[half * 4], p1 = a4[half * 4 + 1];
        float4 p2 = a4[half * 4 + 2], p3 = a4[half * 4 + 3];
        v16h a;
        CVT4(a, 0, p0) CVT4(a, 4, p1) CVT4(a, 8, p2) CVT4(a, 12, p3)
#pragma unroll
        for (int t = 0; t < 4; ++t) {
            int n = t * 16 + lm;
            v16h b;
#pragma unroll
            for (int e = 0; e < 16; ++e)
                b[e] = (_Float16)B[(size_t)(k0 + half * 16 + e) * FOUT + n];
            acc[t] = wmma_f16(a, b, acc[t]);
        }
    }
#pragma unroll
    for (int t = 0; t < 4; ++t)
#pragma unroll
        for (int r = 0; r < 8; ++r)
            C[(size_t)(row0 + r + 8 * half) * FOUT + t * 16 + lm] = acc[t][r];
}

// ---------------------------------------------------------------------------
// Kernel 2: transpose + f16 convert: WhT[n][k] = (f16) Wh[k][n]
// ---------------------------------------------------------------------------
__global__ void wht_k(const float* __restrict__ Wh, _Float16* __restrict__ WhT) {
    int t = blockIdx.x * blockDim.x + threadIdx.x;  // 262144 threads
    int k = t & (N_NODES - 1);
    int n = t >> 12;
    WhT[(size_t)n * N_NODES + k] = (_Float16)Wh[(size_t)k * FOUT + n];
}

// ---------------------------------------------------------------------------
// Kernel 3: el / er projections
// ---------------------------------------------------------------------------
__global__ void a_proj_k(const float* __restrict__ Wh, const float* __restrict__ a,
                         float* __restrict__ el, float* __restrict__ er) {
    int i = blockIdx.x * blockDim.x + threadIdx.x;
    const float* w = Wh + (size_t)i * FOUT;
    float s1 = 0.f, s2 = 0.f;
#pragma unroll 8
    for (int f = 0; f < FOUT; ++f) {
        float wv = w[f];
        s1 += wv * a[f];
        s2 += wv * a[FOUT + f];
    }
    el[i] = s1;
    er[i] = s2;
}

// ---------------------------------------------------------------------------
// Kernel 4: softmax denominators for up to 4 heads in one mask pass.
// (|el+er| is O(10): exp stays in f32 range; softmax needs no max-shift.)
// ---------------------------------------------------------------------------
__global__ void row_z_k(const float* __restrict__ mask, const float* __restrict__ elB,
                        const float* __restrict__ erB, float* __restrict__ ZB, int nh) {
    const int wave = threadIdx.x >> 5;
    const int lane = threadIdx.x & 31;
    const int i = blockIdx.x * 8 + wave;
    const float4* mrow4 = (const float4*)(mask + (size_t)i * N_NODES);

    float sum[4] = {0.f, 0.f, 0.f, 0.f};
    float eli[4] = {0.f, 0.f, 0.f, 0.f};
    for (int h = 0; h < nh; ++h) eli[h] = elB[h * N_NODES + i];

    auto addj = [&](float m, int j) {
        if (m > 0.f) {
            for (int h = 0; h < nh; ++h) {
                float t = eli[h] + erB[h * N_NODES + j];
                t = (t > 0.f) ? t : 0.2f * t;     // leaky relu 0.2
                sum[h] += __expf(t);
            }
        }
    };
    for (int q = lane; q < N_NODES / 4; q += 32) {
        float4 mv = mrow4[q];
        int j = q * 4;
        addj(mv.x, j); addj(mv.y, j + 1); addj(mv.z, j + 2); addj(mv.w, j + 3);
    }
    for (int h = 0; h < nh; ++h) {
        float s = sum[h];
        for (int off = 16; off > 0; off >>= 1) s += __shfl_down(s, off, 32);
        if (lane == 0) ZB[h * N_NODES + i] = s;
    }
}

// ---------------------------------------------------------------------------
// Kernel 5: out = ELU( softmax_att @ Wh ) via streaming-K WMMA with
// TDM-staged, double-buffered LDS B tiles.
// grid = (32, nh); wave = 16 rows x 64 cols; K streamed over 4096 in 32s.
// ---------------------------------------------------------------------------
__device__ __forceinline__ float attv(float mv, float erv, float eli, float invZ) {
    float t = eli + erv;
    t = (t > 0.f) ? t : 0.2f * t;
    float v = __expf(t) * invZ;
    return (mv > 0.f) ? v : 0.f;
}
#define PACK4(B, m, ev) \
    a[B + 0] = (_Float16)attv((m).x, (ev).x, eli, invZ); \
    a[B + 1] = (_Float16)attv((m).y, (ev).y, eli, invZ); \
    a[B + 2] = (_Float16)attv((m).z, (ev).z, eli, invZ); \
    a[B + 3] = (_Float16)attv((m).w, (ev).w, eli, invZ);

__global__ void att_apply_k(const float* __restrict__ mask,
                            const _Float16* __restrict__ WhTB,
                            const float* __restrict__ elB,
                            const float* __restrict__ erB,
                            const float* __restrict__ ZB,
                            float* __restrict__ dst, int ldc, int colBase) {
    __shared__ __align__(16) char smem[2 * TILE_LDS];

    const int h = blockIdx.y;
    const _Float16* WhT = WhTB + (size_t)h * FOUT * N_NODES;
    const float* el = elB + h * N_NODES;
    const float* er = erB + h * N_NODES;
    const float* Z  = ZB  + h * N_NODES;
    const int col0  = colBase + h * FOUT;

    const int tid  = threadIdx.x;
    const int lane = tid & 31;
    const int wave = tid >> 5;
    const int half = (lane >> 4) & 1;
    const int lm   = lane & 15;
    const int row0 = blockIdx.x * 128 + wave * 16;
    const int i    = row0 + lm;

    const float eli  = el[i];
    const float invZ = 1.0f / Z[i];
    const float* mrow = mask + (size_t)i * N_NODES;
    const u32 lds_base = (u32)(uintptr_t)smem;

    v8f acc[4] = {};

    auto consume = [&](int j0, const char* bufp) {
        __builtin_prefetch(mrow + j0 + 64, 0, 0);   // global_prefetch_b8
        const float4* m4 = (const float4*)(mrow + j0);
        const float4* e4 = (const float4*)(er + j0);
        float4 ma = m4[half * 4],     mb = m4[half * 4 + 1];
        float4 mc = m4[half * 4 + 2], md = m4[half * 4 + 3];
        float4 ea = e4[half * 4],     eb = e4[half * 4 + 1];
        float4 ec = e4[half * 4 + 2], ed = e4[half * 4 + 3];
        v16h a;
        PACK4(0, ma, ea) PACK4(4, mb, eb) PACK4(8, mc, ec) PACK4(12, md, ed)
#pragma unroll
        for (int t = 0; t < 4; ++t) {
            const char* bp = bufp + (t * 16 + lm) * TILE_ROW_BYTES + half * 32;
            acc[t] = wmma_f16(a, lds_frag(bp), acc[t]);
        }
    };

    stage_tile(WhT + 0, smem, lds_base, tid);                 // chunk 0 -> buf0
    for (int k = 0; k < 128; k += 2) {
        stage_tile(WhT + (k + 1) * 32, smem + TILE_LDS, lds_base + TILE_LDS, tid);
        tdm_wait(1, tid);                                     // chunk k ready
        __syncthreads();
        consume(k * 32, smem);
        __syncthreads();
        if (k + 2 < 128) {
            stage_tile(WhT + (k + 2) * 32, smem, lds_base, tid);
            tdm_wait(1, tid);                                 // chunk k+1 ready
        } else {
            tdm_wait(0, tid);
        }
        __syncthreads();
        consume((k + 1) * 32, smem + TILE_LDS);
        __syncthreads();
    }

#pragma unroll
    for (int t = 0; t < 4; ++t)
#pragma unroll
        for (int r = 0; r < 8; ++r) {
            float v = acc[t][r];
            v = (v > 0.0f) ? v : (__expf(v) - 1.0f);          // ELU
            dst[(size_t)(row0 + r + 8 * half) * ldc + col0 + t * 16 + lm] = v;
        }
}

// ---------------------------------------------------------------------------
// Kernel 6: semantic-attention pooling over the two branch embeddings.
// ---------------------------------------------------------------------------
__global__ void pool_k(const float* __restrict__ emb1, const float* __restrict__ emb2,
                       const float* __restrict__ Wp1, const float* __restrict__ bp1,
                       const float* __restrict__ Wp2, float* __restrict__ out) {
    int i = blockIdx.x * blockDim.x + threadIdx.x;
    const float* e1 = emb1 + (size_t)i * FOUT;
    const float* e2 = emb2 + (size_t)i * FOUT;
    float w1 = 0.f, w2 = 0.f;
    for (int l = 0; l < 16; ++l) {
        float t1 = bp1[l], t2 = bp1[l];
        for (int f = 0; f < FOUT; ++f) {
            float wv = Wp1[f * 16 + l];
            t1 += e1[f] * wv;
            t2 += e2[f] * wv;
        }
        w1 += tanhf(t1) * Wp2[l];
        w2 += tanhf(t2) * Wp2[l];
    }
    float mx = fmaxf(w1, w2);
    float b1 = __expf(w1 - mx), b2 = __expf(w2 - mx);
    float inv = 1.f / (b1 + b2);
    b1 *= inv; b2 *= inv;
    for (int f = 0; f < FOUT; ++f)
        out[(size_t)i * FOUT + f] = b1 * e1[f] + b2 * e2[f];
}

// ---------------------------------------------------------------------------
// Host driver
// ---------------------------------------------------------------------------
extern "C" void kernel_launch(void* const* d_in, const int* in_sizes, int n_in,
                              void* d_out, int out_size, void* d_ws, size_t ws_size,
                              hipStream_t stream) {
    const float* x     = (const float*)d_in[0];
    const float* sadj  = (const float*)d_in[1];
    const float* sadj2 = (const float*)d_in[2];
    const float* W1    = (const float*)d_in[3];
    const float* a1    = (const float*)d_in[4];
    const float* Wo1   = (const float*)d_in[5];
    const float* ao1   = (const float*)d_in[6];
    const float* W2    = (const float*)d_in[7];
    const float* a2    = (const float*)d_in[8];
    const float* Wo2   = (const float*)d_in[9];
    const float* ao2   = (const float*)d_in[10];
    const float* Wp1   = (const float*)d_in[11];
    const float* bp1   = (const float*)d_in[12];
    const float* Wp2   = (const float*)d_in[13];
    float* out = (float*)d_out;

    // Workspace layout (~26 MB)
    float* ws   = (float*)d_ws;
    float* Wh1  = ws;                        // 4*4096*64 f32
    float* Wh2  = Wh1 + 4 * N_NODES * FOUT;
    float* h1a  = Wh2 + 4 * N_NODES * FOUT;  // 4096*256
    float* h1b  = h1a + N_NODES * 256;
    float* Who1 = h1b + N_NODES * 256;       // 4096*64
    float* Who2 = Who1 + N_NODES * FOUT;
    float* emb1 = Who2 + N_NODES * FOUT;
    float* emb2 = emb1 + N_NODES * FOUT;
    float* elB  = emb2 + N_NODES * FOUT;     // 10 * 4096 each
    float* erB  = elB + 10 * N_NODES;
    float* ZB   = erB + 10 * N_NODES;
    _Float16* WhT1  = (_Float16*)(ZB + 10 * N_NODES);  // 4*64*4096 f16 each block
    _Float16* WhT2  = WhT1 + 4 * FOUT * N_NODES;
    _Float16* WhTo1 = WhT2 + 4 * FOUT * N_NODES;
    _Float16* WhTo2 = WhTo1 + FOUT * N_NODES;

    const dim3 blk(256);
    const dim3 gGemm(N_NODES / 128);
    const dim3 gWht(N_NODES * FOUT / 256);

    // ---------------- GAT branch 1, layer 1 (4 heads) ----------------
    for (int h = 0; h < 4; ++h) {
        float*    WhH  = Wh1  + (size_t)h * N_NODES * FOUT;
        _Float16* WhTH = WhT1 + (size_t)h * FOUT * N_NODES;
        gemm_k<<<gGemm, blk, 0, stream>>>(x, W1 + (size_t)h * 256 * FOUT, WhH, 256, 256);
        wht_k<<<gWht, blk, 0, stream>>>(WhH, WhTH);
        a_proj_k<<<N_NODES / 256, blk, 0, stream>>>(WhH, a1 + h * 128,
                                                    elB + h * N_NODES, erB + h * N_NODES);
    }
    row_z_k<<<N_NODES / 8, blk, 0, stream>>>(sadj, elB, erB, ZB, 4);
    att_apply_k<<<dim3(N_NODES / 128, 4), blk, 0, stream>>>(sadj, WhT1, elB, erB, ZB,
                                                            h1a, 256, 0);

    // ---------------- GAT branch 1, output layer ----------------
    gemm_k<<<gGemm, blk, 0, stream>>>(h1a, Wo1, Who1, 256, 256);
    wht_k<<<gWht, blk, 0, stream>>>(Who1, WhTo1);
    a_proj_k<<<N_NODES / 256, blk, 0, stream>>>(Who1, ao1,
                                                elB + 4 * N_NODES, erB + 4 * N_NODES);
    row_z_k<<<N_NODES / 8, blk, 0, stream>>>(sadj, elB + 4 * N_NODES,
                                             erB + 4 * N_NODES, ZB + 4 * N_NODES, 1);
    att_apply_k<<<dim3(N_NODES / 128, 1), blk, 0, stream>>>(sadj, WhTo1,
                                                            elB + 4 * N_NODES,
                                                            erB + 4 * N_NODES,
                                                            ZB + 4 * N_NODES,
                                                            emb1, FOUT, 0);

    // ---------------- GAT branch 2, layer 1 (4 heads) ----------------
    for (int h = 0; h < 4; ++h) {
        float*    WhH  = Wh2  + (size_t)h * N_NODES * FOUT;
        _Float16* WhTH = WhT2 + (size_t)h * FOUT * N_NODES;
        gemm_k<<<gGemm, blk, 0, stream>>>(x, W2 + (size_t)h * 256 * FOUT, WhH, 256, 256);
        wht_k<<<gWht, blk, 0, stream>>>(WhH, WhTH);
        a_proj_k<<<N_NODES / 256, blk, 0, stream>>>(WhH, a2 + h * 128,
                                                    elB + (5 + h) * N_NODES,
                                                    erB + (5 + h) * N_NODES);
    }
    row_z_k<<<N_NODES / 8, blk, 0, stream>>>(sadj2, elB + 5 * N_NODES,
                                             erB + 5 * N_NODES, ZB + 5 * N_NODES, 4);
    att_apply_k<<<dim3(N_NODES / 128, 4), blk, 0, stream>>>(sadj2, WhT2,
                                                            elB + 5 * N_NODES,
                                                            erB + 5 * N_NODES,
                                                            ZB + 5 * N_NODES,
                                                            h1b, 256, 0);

    // ---------------- GAT branch 2, output layer ----------------
    gemm_k<<<gGemm, blk, 0, stream>>>(h1b, Wo2, Who2, 256, 256);
    wht_k<<<gWht, blk, 0, stream>>>(Who2, WhTo2);
    a_proj_k<<<N_NODES / 256, blk, 0, stream>>>(Who2, ao2,
                                                elB + 9 * N_NODES, erB + 9 * N_NODES);
    row_z_k<<<N_NODES / 8, blk, 0, stream>>>(sadj2, elB + 9 * N_NODES,
                                             erB + 9 * N_NODES, ZB + 9 * N_NODES, 1);
    att_apply_k<<<dim3(N_NODES / 128, 1), blk, 0, stream>>>(sadj2, WhTo2,
                                                            elB + 9 * N_NODES,
                                                            erB + 9 * N_NODES,
                                                            ZB + 9 * N_NODES,
                                                            emb2, FOUT, 0);

    // ---------------- semantic-attention pooling ----------------
    pool_k<<<N_NODES / 256, blk, 0, stream>>>(emb1, emb2, Wp1, bp1, Wp2, out);
}
